// ConvBertForBinaryClassification_18743237279815
// MI455X (gfx1250) — matrix operand order repaired and checked by reading, not compiled, not run on hardware
//
#include <hip/hip_runtime.h>
#include <hip/hip_bf16.h>

#define B_  8
#define S_  1024
#define H_  1536
#define NH_ 8
#define HD_ 96
#define AH_ 768
#define K_  7
#define IM_ 3072
#define M_  (B_ * S_)   // 8192 rows

typedef __attribute__((ext_vector_type(16))) __bf16 bf16x16;
typedef __attribute__((ext_vector_type(8)))  __bf16 bf16x8;
typedef __attribute__((ext_vector_type(8)))  float  floatx8;

// ---------------------------------------------------------------------------
// WMMA fragment builders (layouts per CDNA5 ISA 7.12.2, 16-bit 16x16x32)
// A: lane m=lane&15, half=lane>>4; elem e -> K = (e&8)*2 + half*8 + (e&7)
//    => two contiguous 8-elem (16B) chunks at k0+half*8 and k0+16+half*8
// B: lane n=lane&15, half=lane>>4; elem e -> K = (e&8)*2 + half*8 + (e&7)
//    => two contiguous 8-elem (16B) chunks at k0+half*8 and k0+16+half*8
// C/D f32: VGPR r -> M = r + 8*half, N = lane&15
// ---------------------------------------------------------------------------
__device__ __forceinline__ bf16x16 frag_a_bf16(const __bf16* base, int ld,
                                               int row0, int k0, int lane) {
  int m = lane & 15, half = (lane >> 4) & 1;
  const __bf16* p = base + (size_t)(row0 + m) * ld + k0 + half * 8;
  union { bf16x16 v; bf16x8 h[2]; } u;
  u.h[0] = *(const bf16x8*)(p);
  u.h[1] = *(const bf16x8*)(p + 16);
  return u.v;
}

__device__ __forceinline__ bf16x16 frag_b_bf16(const __bf16* base, int ld,
                                               int col0, int k0, int lane) {
  int n = lane & 15, half = (lane >> 4) & 1;
  const __bf16* p = base + (size_t)(col0 + n) * ld + k0 + half * 8;
  union { bf16x16 v; bf16x8 h[2]; } u;
  u.h[0] = *(const bf16x8*)(p);
  u.h[1] = *(const bf16x8*)(p + 16);
  return u.v;
}

#define SROW 1028   // LDS score row stride (floats): 16B aligned, bank-skewed

__device__ __forceinline__ bf16x16 frag_a_lds(const float* sc, int row0,
                                              int k0, int lane) {
  int m = lane & 15, half = (lane >> 4) & 1;
  const float* p = sc + (size_t)(row0 + m) * SROW + k0 + half * 8;
  float4 f0 = *(const float4*)(p);
  float4 f1 = *(const float4*)(p + 4);
  float4 f2 = *(const float4*)(p + 16);
  float4 f3 = *(const float4*)(p + 20);
  bf16x16 v;
  v[0]  = (__bf16)f0.x; v[1]  = (__bf16)f0.y; v[2]  = (__bf16)f0.z; v[3]  = (__bf16)f0.w;
  v[4]  = (__bf16)f1.x; v[5]  = (__bf16)f1.y; v[6]  = (__bf16)f1.z; v[7]  = (__bf16)f1.w;
  v[8]  = (__bf16)f2.x; v[9]  = (__bf16)f2.y; v[10] = (__bf16)f2.z; v[11] = (__bf16)f2.w;
  v[12] = (__bf16)f3.x; v[13] = (__bf16)f3.y; v[14] = (__bf16)f3.z; v[15] = (__bf16)f3.w;
  return v;
}

// ---------------------------------------------------------------------------
// Generic WMMA GEMM: C[M,N] = A_bf16[M,K] @ WT_bf16[N,K] + bias
// Block: 256 thr = 8 waves in 4(M) x 2(N); each wave owns a 32x32 patch
// (2x2 tiles of 16x16 -> 4 WMMAs per 32-K step, 2 A-frags + 2 B-frags).
// Block tile = 128(M) x 64(N).
// EPI: 0 = f32 out, 1 = bf16 out, 2 = exact GELU then bf16 out.
// ---------------------------------------------------------------------------
template <int EPI>
__device__ __forceinline__ void epi_store(const floatx8& acc, int m0, int n,
                                          float bv, float* Cf, __bf16* Cb,
                                          int Ndim, int half) {
#pragma unroll
  for (int r = 0; r < 8; ++r) {
    int m = m0 + r + half * 8;
    float x = acc[r] + bv;
    if (EPI == 2) x = 0.5f * x * (1.f + erff(x * 0.70710678118f));
    if (EPI == 0) Cf[(size_t)m * Ndim + n] = x;
    else          Cb[(size_t)m * Ndim + n] = (__bf16)x;
  }
}

template <int EPI>
__global__ __launch_bounds__(256) void gemm_bf16_wmma(
    const __bf16* __restrict__ A, const __bf16* __restrict__ WT,
    const float* __restrict__ bias, int bias_n,
    float* __restrict__ Cf, __bf16* __restrict__ Cb, int Ndim, int Kdim) {
  int wave = threadIdx.x >> 5, lane = threadIdx.x & 31;
  int row0 = blockIdx.x * 128 + (wave & 3) * 32;
  int col0 = blockIdx.y * 64 + (wave >> 2) * 32;
  int half = (lane >> 4) & 1;
  const __bf16* pa0 = A  + (size_t)(row0 + (lane & 15)) * Kdim + half * 8;
  const __bf16* pa1 = pa0 + (size_t)16 * Kdim;
  const __bf16* pb0 = WT + (size_t)(col0 + (lane & 15)) * Kdim + half * 8;
  const __bf16* pb1 = pb0 + (size_t)16 * Kdim;
  floatx8 acc00 = {}, acc01 = {}, acc10 = {}, acc11 = {};
  for (int k0 = 0; k0 < Kdim; k0 += 32) {
    if (k0 + 256 < Kdim) {
      __builtin_prefetch(pa0 + k0 + 256, 0, 0);
      __builtin_prefetch(pa1 + k0 + 256, 0, 0);
      __builtin_prefetch(pb0 + k0 + 256, 0, 0);
      __builtin_prefetch(pb1 + k0 + 256, 0, 0);
    }
    union { bf16x16 v; bf16x8 h[2]; } a0, a1, b0, b1;
    a0.h[0] = *(const bf16x8*)(pa0 + k0);
    a0.h[1] = *(const bf16x8*)(pa0 + k0 + 16);
    a1.h[0] = *(const bf16x8*)(pa1 + k0);
    a1.h[1] = *(const bf16x8*)(pa1 + k0 + 16);
    b0.h[0] = *(const bf16x8*)(pb0 + k0);
    b0.h[1] = *(const bf16x8*)(pb0 + k0 + 16);
    b1.h[0] = *(const bf16x8*)(pb1 + k0);
    b1.h[1] = *(const bf16x8*)(pb1 + k0 + 16);
    acc00 = __builtin_amdgcn_wmma_f32_16x16x32_bf16(
        false, a0.v, false, b0.v, (short)0, acc00, false, false);
    acc01 = __builtin_amdgcn_wmma_f32_16x16x32_bf16(
        false, a0.v, false, b1.v, (short)0, acc01, false, false);
    acc10 = __builtin_amdgcn_wmma_f32_16x16x32_bf16(
        false, a1.v, false, b0.v, (short)0, acc10, false, false);
    acc11 = __builtin_amdgcn_wmma_f32_16x16x32_bf16(
        false, a1.v, false, b1.v, (short)0, acc11, false, false);
  }
  int n0 = col0 + (lane & 15), n1 = n0 + 16;
  float bv0 = (bias && n0 < bias_n) ? bias[n0] : 0.f;
  float bv1 = (bias && n1 < bias_n) ? bias[n1] : 0.f;
  epi_store<EPI>(acc00, row0,      n0, bv0, Cf, Cb, Ndim, half);
  epi_store<EPI>(acc01, row0,      n1, bv1, Cf, Cb, Ndim, half);
  epi_store<EPI>(acc10, row0 + 16, n0, bv0, Cf, Cb, Ndim, half);
  epi_store<EPI>(acc11, row0 + 16, n1, bv1, Cf, Cb, Ndim, half);
}

// ---------------------------------------------------------------------------
// Fused attention per (b, head, 32-query block): scores -> LDS, softmax, @V.
// ---------------------------------------------------------------------------
__global__ __launch_bounds__(256) void attention_kernel(
    const __bf16* __restrict__ q_bf, const __bf16* __restrict__ k_bf,
    const __bf16* __restrict__ vT_bf, __bf16* __restrict__ ctx_cat) {
  __shared__ float sc[32 * SROW];        // 131,584 B
  __shared__ float redm[128], reds[128];
  int bh = blockIdx.x, b = bh >> 3, h = bh & 7;
  int q_row = b * S_ + blockIdx.y * 32;  // global activation row
  int wave = threadIdx.x >> 5, lane = threadIdx.x & 31, tid = threadIdx.x;
  const __bf16* qh = q_bf + (size_t)q_row * AH_ + h * HD_;
  const __bf16* kh = k_bf + (size_t)b * S_ * AH_ + h * HD_;
  const __bf16* vh = vT_bf + (size_t)bh * HD_ * S_;
  const float scale = 0.1020620726f;     // 1/sqrt(96)

  // scores[32,1024]: 2 q-tiles x 64 k-tiles, K = 96 (3 WMMA steps)
  for (int t = wave; t < 128; t += 8) {
    int qt = t & 1, kt = t >> 1;
    floatx8 acc = {};
#pragma unroll
    for (int d0 = 0; d0 < HD_; d0 += 32) {
      bf16x16 a  = frag_a_bf16(qh, AH_, qt * 16, d0, lane);
      bf16x16 bb = frag_b_bf16(kh, AH_, kt * 16, d0, lane);
      acc = __builtin_amdgcn_wmma_f32_16x16x32_bf16(
          false, a, false, bb, (short)0, acc, false, false);
    }
    int n = lane & 15, half = (lane >> 4) & 1;
#pragma unroll
    for (int r = 0; r < 8; ++r)
      sc[(qt * 16 + r + half * 8) * SROW + kt * 16 + n] = acc[r] * scale;
  }
  __syncthreads();

  // softmax over 1024 keys, 4 threads per row
  int r = tid >> 2, part = tid & 3;
  if (tid < 128) {
    float* row = sc + (size_t)r * SROW;
    float mx = -3.0e38f;
    for (int i = part * 256; i < part * 256 + 256; ++i) mx = fmaxf(mx, row[i]);
    redm[tid] = mx;
  }
  __syncthreads();
  if (tid < 128) {
    float rowmax = fmaxf(fmaxf(redm[r * 4], redm[r * 4 + 1]),
                         fmaxf(redm[r * 4 + 2], redm[r * 4 + 3]));
    float* row = sc + (size_t)r * SROW;
    float s = 0.f;
    for (int i = part * 256; i < part * 256 + 256; ++i) {
      float e = expf(row[i] - rowmax); row[i] = e; s += e;
    }
    reds[tid] = s;
  }
  __syncthreads();
  if (tid < 128) {
    float inv = 1.f / (reds[r * 4] + reds[r * 4 + 1] +
                       reds[r * 4 + 2] + reds[r * 4 + 3]);
    float* row = sc + (size_t)r * SROW;
    for (int i = part * 256; i < part * 256 + 256; ++i) row[i] *= inv;
  }
  __syncthreads();

  // ctx[32,96] = probs @ V : 2 x 6 tiles, K = 1024
  for (int t = wave; t < 12; t += 8) {
    int qt = t & 1, dt = t >> 1;
    floatx8 acc = {};
    for (int k0 = 0; k0 < S_; k0 += 32) {
      bf16x16 a  = frag_a_lds(sc, qt * 16, k0, lane);
      bf16x16 bb = frag_b_bf16(vh, S_, dt * 16, k0, lane);
      acc = __builtin_amdgcn_wmma_f32_16x16x32_bf16(
          false, a, false, bb, (short)0, acc, false, false);
    }
    int n = lane & 15, half = (lane >> 4) & 1;
#pragma unroll
    for (int rr = 0; rr < 8; ++rr) {
      int row_g = q_row + qt * 16 + rr + half * 8;
      ctx_cat[(size_t)row_g * H_ + h * HD_ + dt * 16 + n] = (__bf16)acc[rr];
    }
  }
}

// ---------------------------------------------------------------------------
// Elementwise / reduction helpers
// ---------------------------------------------------------------------------
__global__ void cvt_f32_to_bf16(const float* __restrict__ s,
                                __bf16* __restrict__ d, size_t n) {
  size_t i = (size_t)blockIdx.x * 256 + threadIdx.x;
  if (i < n) d[i] = (__bf16)s[i];
}

// src [K,N] f32 -> dst [Npad,K] bf16 (zero-padded rows)
__global__ void transpose_cvt(const float* __restrict__ src,
                              __bf16* __restrict__ dst, int K, int N, int Npad) {
  size_t i = (size_t)blockIdx.x * 256 + threadIdx.x;
  if (i >= (size_t)K * Npad) return;
  int n = (int)(i % Npad), k = (int)(i / Npad);
  dst[(size_t)n * K + k] = (n < N) ? (__bf16)src[(size_t)k * N + n] : (__bf16)0.f;
}

// v f32 [M,AH] -> vT bf16 [B,NH,HD,S]
__global__ void build_vT(const float* __restrict__ v, __bf16* __restrict__ vT) {
  size_t i = (size_t)blockIdx.x * 256 + threadIdx.x;
  if (i >= (size_t)B_ * NH_ * HD_ * S_) return;
  int s = (int)(i & (S_ - 1));
  size_t rr = i >> 10;
  int d = (int)(rr % HD_), bh = (int)(rr / HD_);
  int b = bh >> 3, h = bh & 7;
  vT[i] = (__bf16)v[((size_t)(b * S_ + s)) * AH_ + h * HD_ + d];
}

// depthwise conv k=7, 'same' padding
__global__ void dwconv(const float* __restrict__ emb, const float* __restrict__ dw,
                       __bf16* __restrict__ out) {
  size_t i = (size_t)blockIdx.x * 256 + threadIdx.x;
  if (i >= (size_t)M_ * H_) return;
  int c = (int)(i % H_); size_t m = i / H_;
  int s = (int)(m & (S_ - 1));
  float acc = 0.f;
#pragma unroll
  for (int j = 0; j < K_; ++j) {
    int ss = s + j - 3;
    if (ss >= 0 && ss < S_) acc += emb[(m + j - 3) * H_ + c] * dw[c * K_ + j];
  }
  out[i] = (__bf16)acc;
}

__global__ void mul_to_bf16(const float* __restrict__ a, const float* __restrict__ b,
                            __bf16* __restrict__ o, size_t n) {
  size_t i = (size_t)blockIdx.x * 256 + threadIdx.x;
  if (i < n) o[i] = (__bf16)(a[i] * b[i]);
}

// softmax over the 7 span-kernel taps per (row, head); ck stored [M,64]
__global__ void ck_softmax(const float* __restrict__ ck, float* __restrict__ o) {
  int i = blockIdx.x * 256 + threadIdx.x;
  if (i >= M_ * NH_) return;
  int h = i & 7, m = i >> 3;
  const float* p = ck + (size_t)m * 64 + h * 7;
  float mx = p[0];
#pragma unroll
  for (int j = 1; j < K_; ++j) mx = fmaxf(mx, p[j]);
  float e[K_], s = 0.f;
#pragma unroll
  for (int j = 0; j < K_; ++j) { e[j] = expf(p[j] - mx); s += e[j]; }
  float inv = 1.f / s;
  float* q = o + (size_t)m * 64 + h * 7;
#pragma unroll
  for (int j = 0; j < K_; ++j) q[j] = e[j] * inv;
}

// dynamic conv heads -> second half of ctx_cat
__global__ void conv_out_k(const float* __restrict__ cksm, const float* __restrict__ co,
                           __bf16* __restrict__ ctx_cat) {
  size_t i = (size_t)blockIdx.x * 256 + threadIdx.x;
  if (i >= (size_t)M_ * AH_) return;
  int c = (int)(i % AH_); size_t m = i / AH_;
  int h = c / HD_, s = (int)(m & (S_ - 1));
  float acc = 0.f;
#pragma unroll
  for (int j = 0; j < K_; ++j) {
    int ss = s + j - 3;
    if (ss >= 0 && ss < S_)
      acc += cksm[m * 64 + h * 7 + j] * co[(m + j - 3) * AH_ + c];
  }
  ctx_cat[m * H_ + AH_ + c] = (__bf16)acc;
}

__global__ __launch_bounds__(256) void ln_residual(
    const float* __restrict__ x, const float* __restrict__ resid,
    const float* __restrict__ g, const float* __restrict__ bta,
    float* __restrict__ outf, __bf16* __restrict__ outb, int Hdim) {
  __shared__ float red[256];
  size_t base = (size_t)blockIdx.x * Hdim;
  int tid = threadIdx.x;
  float s = 0.f;
  for (int i = tid; i < Hdim; i += 256) s += x[base + i] + resid[base + i];
  red[tid] = s; __syncthreads();
  for (int o = 128; o > 0; o >>= 1) { if (tid < o) red[tid] += red[tid + o]; __syncthreads(); }
  float mean = red[0] / Hdim;
  __syncthreads();
  float vv = 0.f;
  for (int i = tid; i < Hdim; i += 256) {
    float d = x[base + i] + resid[base + i] - mean; vv += d * d;
  }
  red[tid] = vv; __syncthreads();
  for (int o = 128; o > 0; o >>= 1) { if (tid < o) red[tid] += red[tid + o]; __syncthreads(); }
  float rstd = rsqrtf(red[0] / Hdim + 1e-12f);
  for (int i = tid; i < Hdim; i += 256) {
    float y = (x[base + i] + resid[base + i] - mean) * rstd * g[i] + bta[i];
    outf[base + i] = y;
    if (outb) outb[base + i] = (__bf16)y;
  }
}

__global__ __launch_bounds__(256) void pool_decode(
    const float* __restrict__ lo, const float* __restrict__ wd,
    const float* __restrict__ bd, float* __restrict__ out) {
  __shared__ float red[256];
  int b = blockIdx.x, tid = threadIdx.x;
  float acc = 0.f;
  for (int hh = tid; hh < H_; hh += 256) {
    const float* p = lo + (size_t)b * S_ * H_ + hh;
    float mx = -3.0e38f;
    for (int s = 0; s < S_; ++s) mx = fmaxf(mx, p[(size_t)s * H_]);
    acc += mx * wd[hh];
  }
  red[tid] = acc; __syncthreads();
  for (int o = 128; o > 0; o >>= 1) { if (tid < o) red[tid] += red[tid + o]; __syncthreads(); }
  if (tid == 0) out[b] = red[0] + bd[0];
}

// ---------------------------------------------------------------------------
extern "C" void kernel_launch(void* const* d_in, const int* in_sizes, int n_in,
                              void* d_out, int out_size, void* d_ws, size_t ws_size,
                              hipStream_t stream) {
  (void)in_sizes; (void)n_in; (void)out_size; (void)ws_size;
  const float* embed = (const float*)d_in[0];
  const float* wq  = (const float*)d_in[1];  const float* bq  = (const float*)d_in[2];
  const float* wk  = (const float*)d_in[3];  const float* bk  = (const float*)d_in[4];
  const float* wv  = (const float*)d_in[5];  const float* bv  = (const float*)d_in[6];
  const float* dw  = (const float*)d_in[7];
  const float* pw  = (const float*)d_in[8];  const float* sep_b = (const float*)d_in[9];
  const float* wck = (const float*)d_in[10]; const float* bck = (const float*)d_in[11];
  const float* wco = (const float*)d_in[12]; const float* bco = (const float*)d_in[13];
  const float* wso = (const float*)d_in[14]; const float* bso = (const float*)d_in[15];
  const float* ln1g = (const float*)d_in[16]; const float* ln1b = (const float*)d_in[17];
  const float* wi  = (const float*)d_in[18]; const float* bi  = (const float*)d_in[19];
  const float* wo  = (const float*)d_in[20]; const float* bo  = (const float*)d_in[21];
  const float* ln2g = (const float*)d_in[22]; const float* ln2b = (const float*)d_in[23];
  const float* wd  = (const float*)d_in[24]; const float* bd  = (const float*)d_in[25];

  char* ws = (char*)d_ws;
  size_t off = 0;
  auto alloc = [&](size_t bytes) -> void* {
    void* p = ws + off; off += (bytes + 255) & ~(size_t)255; return p;
  };
  __bf16* embed_bf = (__bf16*)alloc((size_t)M_ * H_ * 2);
  __bf16* wqT  = (__bf16*)alloc((size_t)AH_ * H_ * 2);
  __bf16* wkT  = (__bf16*)alloc((size_t)AH_ * H_ * 2);
  __bf16* wvT  = (__bf16*)alloc((size_t)AH_ * H_ * 2);
  __bf16* pwT  = (__bf16*)alloc((size_t)AH_ * H_ * 2);
  __bf16* wcoT = (__bf16*)alloc((size_t)AH_ * H_ * 2);
  __bf16* wckT = (__bf16*)alloc((size_t)64 * AH_ * 2);
  __bf16* wsoT = (__bf16*)alloc((size_t)H_ * H_ * 2);
  __bf16* wiT  = (__bf16*)alloc((size_t)IM_ * H_ * 2);
  __bf16* woT  = (__bf16*)alloc((size_t)H_ * IM_ * 2);
  float*  q_f  = (float*)alloc((size_t)M_ * AH_ * 4);   // + v_f contiguous:
  float*  v_f  = (float*)alloc((size_t)M_ * AH_ * 4);   //   reused as so/ffo
  float*  co_f = (float*)alloc((size_t)M_ * AH_ * 4);   // + sep contiguous:
  float*  sep_f= (float*)alloc((size_t)M_ * AH_ * 4);   //   reused as attn_out
  __bf16* q_bf = (__bf16*)alloc((size_t)M_ * AH_ * 2);
  __bf16* k_bf = (__bf16*)alloc((size_t)M_ * AH_ * 2);
  __bf16* vT_bf= (__bf16*)alloc((size_t)M_ * AH_ * 2);
  __bf16* dconv_bf = (__bf16*)alloc((size_t)M_ * H_ * 2);
  __bf16* cattn_bf = (__bf16*)alloc((size_t)M_ * AH_ * 2);
  float*  ck   = (float*)alloc((size_t)M_ * 64 * 4);
  float*  cksm = (float*)alloc((size_t)M_ * 64 * 4);
  __bf16* ctx_cat = (__bf16*)alloc((size_t)M_ * H_ * 2);
  __bf16* aout_bf = (__bf16*)alloc((size_t)M_ * H_ * 2);
  __bf16* inter_bf = (__bf16*)alloc((size_t)M_ * IM_ * 2);
  // lifetime-based aliases (stream-ordered, no overlap in time):
  float* so_f   = q_f;                 // spans q_f+v_f  (50.3 MB)
  float* aout_f = co_f;                // spans co_f+sep (50.3 MB)
  float* ffo_f  = q_f;                 // after LN1 consumed so_f
  float* lout_f = (float*)inter_bf;    // after FFN2 consumed inter

  auto g1 = [](size_t n) { return dim3((unsigned)((n + 255) / 256)); };

  // 1) precision conversion + weight transposes
  cvt_f32_to_bf16<<<g1((size_t)M_ * H_), 256, 0, stream>>>(embed, embed_bf, (size_t)M_ * H_);
  transpose_cvt<<<g1((size_t)H_ * AH_), 256, 0, stream>>>(wq,  wqT,  H_, AH_, AH_);
  transpose_cvt<<<g1((size_t)H_ * AH_), 256, 0, stream>>>(wk,  wkT,  H_, AH_, AH_);
  transpose_cvt<<<g1((size_t)H_ * AH_), 256, 0, stream>>>(wv,  wvT,  H_, AH_, AH_);
  transpose_cvt<<<g1((size_t)H_ * AH_), 256, 0, stream>>>(pw,  pwT,  H_, AH_, AH_);
  transpose_cvt<<<g1((size_t)H_ * AH_), 256, 0, stream>>>(wco, wcoT, H_, AH_, AH_);
  transpose_cvt<<<g1((size_t)AH_ * 64), 256, 0, stream>>>(wck, wckT, AH_, NH_ * K_, 64);
  transpose_cvt<<<g1((size_t)H_ * H_),  256, 0, stream>>>(wso, wsoT, H_, H_, H_);
  transpose_cvt<<<g1((size_t)H_ * IM_), 256, 0, stream>>>(wi,  wiT,  H_, IM_, IM_);
  transpose_cvt<<<g1((size_t)IM_ * H_), 256, 0, stream>>>(wo,  woT,  IM_, H_, H_);

  // 2) projections  (block tile 128x64)
  dim3 gp(M_ / 128, AH_ / 64);
  gemm_bf16_wmma<0><<<gp, 256, 0, stream>>>(embed_bf, wqT, bq, AH_, q_f, nullptr, AH_, H_);
  gemm_bf16_wmma<1><<<gp, 256, 0, stream>>>(embed_bf, wkT, bk, AH_, nullptr, k_bf, AH_, H_);
  gemm_bf16_wmma<0><<<gp, 256, 0, stream>>>(embed_bf, wvT, bv, AH_, v_f, nullptr, AH_, H_);
  gemm_bf16_wmma<0><<<gp, 256, 0, stream>>>(embed_bf, wcoT, bco, AH_, co_f, nullptr, AH_, H_);
  cvt_f32_to_bf16<<<g1((size_t)M_ * AH_), 256, 0, stream>>>(q_f, q_bf, (size_t)M_ * AH_);
  build_vT<<<g1((size_t)M_ * AH_), 256, 0, stream>>>(v_f, vT_bf);

  // 3) separable conv + span dynamic conv
  dwconv<<<g1((size_t)M_ * H_), 256, 0, stream>>>(embed, dw, dconv_bf);
  gemm_bf16_wmma<0><<<gp, 256, 0, stream>>>(dconv_bf, pwT, sep_b, AH_, sep_f, nullptr, AH_, H_);
  mul_to_bf16<<<g1((size_t)M_ * AH_), 256, 0, stream>>>(sep_f, q_f, cattn_bf, (size_t)M_ * AH_);
  dim3 gck(M_ / 128, 1);
  gemm_bf16_wmma<0><<<gck, 256, 0, stream>>>(cattn_bf, wckT, bck, NH_ * K_, ck, nullptr, 64, AH_);
  ck_softmax<<<g1((size_t)M_ * NH_), 256, 0, stream>>>(ck, cksm);
  conv_out_k<<<g1((size_t)M_ * AH_), 256, 0, stream>>>(cksm, co_f, ctx_cat);

  // 4) self-attention (fused per (b,h,q-block))
  attention_kernel<<<dim3(B_ * NH_, S_ / 32), 256, 0, stream>>>(q_bf, k_bf, vT_bf, ctx_cat);

  // 5) self-output + LN1
  dim3 gso(M_ / 128, H_ / 64);
  gemm_bf16_wmma<0><<<gso, 256, 0, stream>>>(ctx_cat, wsoT, bso, H_, so_f, nullptr, H_, H_);
  ln_residual<<<M_, 256, 0, stream>>>(so_f, embed, ln1g, ln1b, aout_f, aout_bf, H_);

  // 6) FFN + LN2
  dim3 gi(M_ / 128, IM_ / 64);
  gemm_bf16_wmma<2><<<gi, 256, 0, stream>>>(aout_bf, wiT, bi, IM_, nullptr, inter_bf, IM_, H_);
  gemm_bf16_wmma<0><<<gso, 256, 0, stream>>>(inter_bf, woT, bo, H_, ffo_f, nullptr, H_, IM_);
  ln_residual<<<M_, 256, 0, stream>>>(ffo_f, aout_f, ln2g, ln2b, lout_f, nullptr, H_);

  // 7) max-pool over sequence + decoder
  pool_decode<<<B_, 256, 0, stream>>>(lout_f, wd, bd, (float*)d_out);
}
